// EnhancedGatedFusion_41120016891972
// MI455X (gfx1250) — matrix operand order repaired and checked by reading, not compiled, not run on hardware
//
#include <hip/hip_runtime.h>
#include <hip/hip_bf16.h>

// ---------------------------------------------------------------------------
// Problem constants (from reference): B=4, S=2048 -> T=8192 tokens, D=2048, E=8
// ---------------------------------------------------------------------------
#define TOKENS 8192
#define DIM    2048
#define NEXP   8

// GEMM tiling: block tile 128x128, K-step 32, 8 waves each owning 64x32.
// LDS rows padded to 40 halves (80B) so fragment b128 loads stay 16B-aligned.
#define BM 128
#define BN 128
#define BK 32
#define LSTR 40

typedef __bf16 bf16_t;
typedef bf16_t v16bf __attribute__((ext_vector_type(16)));
typedef float  v8f   __attribute__((ext_vector_type(8)));
typedef unsigned int u32x4 __attribute__((ext_vector_type(4)));
typedef unsigned int u32x4e __attribute__((ext_vector_type(4)));
typedef int i32x4e __attribute__((ext_vector_type(4)));
typedef int i32x8e __attribute__((ext_vector_type(8)));

union Frag {            // one WMMA A/B operand: 8 VGPRs = 32 bytes per lane
    u32x4 q[2];
    v16bf v;
};

// ---------------------------------------------------------------------------
// Feature detection: TDM > async-to-LDS > manual staging
// ---------------------------------------------------------------------------
#if defined(__has_builtin)
#if __has_builtin(__builtin_amdgcn_tensor_load_to_lds) && \
    __has_builtin(__builtin_amdgcn_s_wait_tensorcnt)
#define HAVE_TDM 1
#if __has_include(<hip/amd_detail/amd_gfx1250_TDM.h>)
#define TDM_6ARG 1            /* therock toolchain: 6-arg builtin */
#endif
#endif
#if __has_builtin(__builtin_amdgcn_global_load_async_to_lds_b128)
#define HAVE_ASYNC_LDS 1
#endif
#if __has_builtin(__builtin_amdgcn_s_wait_asynccnt)
#define HAVE_WAIT_ASYNC 1
#endif
#endif

#ifdef HAVE_ASYNC_LDS
typedef int v4i_t __attribute__((vector_size(16)));
typedef __attribute__((address_space(1))) v4i_t* g_v4i_p;
typedef __attribute__((address_space(3))) v4i_t* l_v4i_p;

__device__ __forceinline__ void async_copy_b128(const void* g, void* l, int off16) {
    g_v4i_p gp = (g_v4i_p)(unsigned long long)(uintptr_t)g;
    l_v4i_p lp = (l_v4i_p)(unsigned int)(uintptr_t)l;
    if (off16 == 0)
        __builtin_amdgcn_global_load_async_to_lds_b128(gp, lp, 0, 0);
    else
        __builtin_amdgcn_global_load_async_to_lds_b128(gp, lp, 16, 0);
}
#endif

#ifdef HAVE_TDM
// Issue one TDM descriptor copying a BMxBK (rows x halves) tile from a
// row-major [*, DIM] bf16 matrix into LDS with the padded LSTR layout.
// D# bit layout per CDNA5 ISA 8.3/8.4:
//   g0: count=1 | lds_addr | global_addr[56:0] | type=2
//   g1: data_size=2B, pad_enable, pad_interval=3 (16 DW), pad_amount=3 (4 DW),
//       tensor_dim0/1 = 2048, tile_dim0 = 32, tile_dim1 = 128, stride0 = 2048
__device__ __forceinline__ void tdm_load_tile(const unsigned short* g,
                                              unsigned short* l) {
    const unsigned long long ga = (unsigned long long)(uintptr_t)g;
    const unsigned int la = (unsigned int)(uintptr_t)l;   // LDS byte offset
    u32x4e g0;
    g0[0] = 1u;                                           // count=1 (valid)
    g0[1] = la;                                           // lds_addr
    g0[2] = (unsigned int)ga;                             // global_addr lo
    g0[3] = (unsigned int)((ga >> 32) & 0x1FFFFFFull) | (2u << 30); // hi|type=2
    i32x8e g1;
    g1[0] = (int)((1u << 16) | (1u << 20) | (3u << 22) | (3u << 25));
    g1[1] = (int)(2048u << 16);       // tensor_dim0[15:0] in bits[63:48]
    g1[2] = (int)(2048u << 16);       // tensor_dim1[15:0] in bits[95:80]
    g1[3] = (int)(32u << 16);         // tile_dim0 = 32 elements (bits 127:112)
    g1[4] = 128;                      // tile_dim1 = 128 rows (bits 143:128)
    g1[5] = 2048;                     // tensor_dim0_stride lo (bits 191:160)
    g1[6] = 0;
    g1[7] = 0;
    i32x4e z4 = {0, 0, 0, 0};
#ifdef TDM_6ARG
    i32x8e z8 = {0, 0, 0, 0, 0, 0, 0, 0};
    __builtin_amdgcn_tensor_load_to_lds(g0, g1, z4, z4, z8, 0);
#else
    __builtin_amdgcn_tensor_load_to_lds(g0, g1, z4, z4, 0);
#endif
}
#endif

// ---------------------------------------------------------------------------
// Staging abstraction (both tiles are BM rows x BK halves, K-contiguous)
// ---------------------------------------------------------------------------
__device__ __forceinline__ void stage_tiles(const unsigned short* gA,
                                            const unsigned short* gB,
                                            unsigned short* lA,
                                            unsigned short* lB, int tid) {
#if defined(HAVE_TDM)
    if ((tid >> 5) == 0) {
        tdm_load_tile(gA, lA);
        tdm_load_tile(gB, lB);
    }
#else
    const int srow = tid >> 1, scol = (tid & 1) * 16;
    const unsigned short* ga = gA + (size_t)srow * DIM + scol;
    const unsigned short* gb = gB + (size_t)srow * DIM + scol;
#if defined(HAVE_ASYNC_LDS)
    async_copy_b128(ga, lA + srow * LSTR + scol, 0);
    async_copy_b128(ga, lA + srow * LSTR + scol, 16);
    async_copy_b128(gb, lB + srow * LSTR + scol, 0);
    async_copy_b128(gb, lB + srow * LSTR + scol, 16);
#else
    const u32x4 a0 = ((const u32x4*)ga)[0];
    const u32x4 a1 = ((const u32x4*)ga)[1];
    const u32x4 b0 = ((const u32x4*)gb)[0];
    const u32x4 b1 = ((const u32x4*)gb)[1];
    *(u32x4*)(lA + srow * LSTR + scol)     = a0;
    *(u32x4*)(lA + srow * LSTR + scol + 8) = a1;
    *(u32x4*)(lB + srow * LSTR + scol)     = b0;
    *(u32x4*)(lB + srow * LSTR + scol + 8) = b1;
#endif
#endif
}

__device__ __forceinline__ void stage_wait(int tid) {
#if defined(HAVE_TDM)
    if ((tid >> 5) == 0) __builtin_amdgcn_s_wait_tensorcnt(0);
#elif defined(HAVE_ASYNC_LDS)
#ifdef HAVE_WAIT_ASYNC
    __builtin_amdgcn_s_wait_asynccnt(0);
#else
    asm volatile("s_wait_asynccnt 0x0" ::: "memory");
#endif
#endif
}

__device__ __forceinline__ unsigned short f32_to_bf16(float f) {
    union { float f; unsigned int u; } c; c.f = f;
    unsigned int u = c.u;
    u += 0x7FFFu + ((u >> 16) & 1u);        // round-to-nearest-even
    return (unsigned short)(u >> 16);
}

// ---------------------------------------------------------------------------
// Transposing fp32 -> bf16 conversion: out[c][r] = bf16(in[r][c]) per matrix.
// Weights land K-contiguous per output column -> no per-tile transpose.
// ---------------------------------------------------------------------------
__global__ __launch_bounds__(256)
void transpose_cvt_kernel(const float* __restrict__ in,
                          unsigned short* __restrict__ out) {
    __shared__ float tile[32][33];
    const int m = blockIdx.z;
    const float* src = in + (size_t)m * DIM * DIM;
    unsigned short* dst = out + (size_t)m * DIM * DIM;
    const int r0 = blockIdx.y * 32, c0 = blockIdx.x * 32;
    const int tx = threadIdx.x, ty = threadIdx.y;
#pragma unroll
    for (int i = 0; i < 4; ++i)
        tile[ty + 8 * i][tx] = src[(size_t)(r0 + ty + 8 * i) * DIM + c0 + tx];
    __syncthreads();
#pragma unroll
    for (int i = 0; i < 4; ++i)
        dst[(size_t)(c0 + ty + 8 * i) * DIM + r0 + tx] =
            f32_to_bf16(tile[tx][ty + 8 * i]);
}

// ---------------------------------------------------------------------------
// Router: logits = x@Wr + br, top-2, softmax -> dense gate matrix G[T,8]
// Also converts x rows to bf16 (xb). One block per token.
// ---------------------------------------------------------------------------
__global__ __launch_bounds__(256)
void router_kernel(const float* __restrict__ x, const float* __restrict__ Wr,
                   const float* __restrict__ br,
                   unsigned short* __restrict__ xb, float* __restrict__ G) {
    const int t   = blockIdx.x;
    const int tid = threadIdx.x;
    const float* xr = x + (size_t)t * DIM;

    float p[NEXP];
#pragma unroll
    for (int e = 0; e < NEXP; ++e) p[e] = 0.f;

#pragma unroll
    for (int i = 0; i < DIM / 256; ++i) {
        const int d = tid + i * 256;
        const float xv = xr[d];
        xb[(size_t)t * DIM + d] = f32_to_bf16(xv);
        const float* w = Wr + (size_t)d * NEXP;
#pragma unroll
        for (int e = 0; e < NEXP; ++e) p[e] += xv * w[e];
    }

    __shared__ float red[NEXP * 256];
#pragma unroll
    for (int e = 0; e < NEXP; ++e) red[e * 256 + tid] = p[e];
    __syncthreads();
    for (int s = 128; s > 0; s >>= 1) {
        if (tid < s) {
#pragma unroll
            for (int e = 0; e < NEXP; ++e)
                red[e * 256 + tid] += red[e * 256 + tid + s];
        }
        __syncthreads();
    }

    if (tid == 0) {
        float lg[NEXP];
#pragma unroll
        for (int e = 0; e < NEXP; ++e) lg[e] = red[e * 256] + br[e];
        int i0 = 0;
#pragma unroll
        for (int e = 1; e < NEXP; ++e) if (lg[e] > lg[i0]) i0 = e;
        int i1 = (i0 == 0) ? 1 : 0;
#pragma unroll
        for (int e = 0; e < NEXP; ++e)
            if (e != i0 && lg[e] > lg[i1]) i1 = e;
        const float m  = fmaxf(lg[i0], lg[i1]);
        const float e0 = __expf(lg[i0] - m);
        const float e1 = __expf(lg[i1] - m);
        const float inv = 1.f / (e0 + e1);
        float g[NEXP];
#pragma unroll
        for (int e = 0; e < NEXP; ++e) g[e] = 0.f;
        g[i0] = e0 * inv;
        g[i1] = e1 * inv;
#pragma unroll
        for (int e = 0; e < NEXP; ++e) G[(size_t)t * NEXP + e] = g[e];
    }
}

// ---------------------------------------------------------------------------
// Expert GEMM: combined[t,:] = sum_e G[t,e] * silu(xb[t,:] @ We[e] + be[e])
// WeT layout: [e][n][k]. Double-buffered, pipelined across K-steps & experts.
// ---------------------------------------------------------------------------
__global__ __launch_bounds__(256)
void expert_gemm_kernel(const unsigned short* __restrict__ xb,
                        const unsigned short* __restrict__ WeT,
                        const float* __restrict__ be,
                        const float* __restrict__ G,
                        unsigned short* __restrict__ comb) {
    __shared__ unsigned short At[2][BM * LSTR];
    __shared__ unsigned short Bt[2][BN * LSTR];

    const int tid  = threadIdx.x;
    const int lane = tid & 31;
    const int wid  = tid >> 5;
    const int lm   = lane & 15;
    const int hi   = (lane >= 16) ? 1 : 0;
    const int m0w  = (wid & 1) * 64;
    const int n0w  = (wid >> 1) * 32;
    const int blockM = blockIdx.x * BM;
    const int blockN = blockIdx.y * BN;

    v8f comb_acc[4][2];
#pragma unroll
    for (int mt = 0; mt < 4; ++mt)
#pragma unroll
        for (int nt = 0; nt < 2; ++nt)
#pragma unroll
            for (int r = 0; r < 8; ++r) comb_acc[mt][nt][r] = 0.f;

    const unsigned short* Ab = xb + (size_t)blockM * DIM;     // A rows base
    // prologue: stage (e=0, k0=0) into buffer 0
    stage_tiles(Ab, WeT + (size_t)blockN * DIM, At[0], Bt[0], tid);
    int buf = 0;

    for (int e = 0; e < NEXP; ++e) {
        v8f acc[4][2];
#pragma unroll
        for (int mt = 0; mt < 4; ++mt)
#pragma unroll
            for (int nt = 0; nt < 2; ++nt)
#pragma unroll
                for (int r = 0; r < 8; ++r) acc[mt][nt][r] = 0.f;

        for (int k0 = 0; k0 < DIM; k0 += BK) {
            stage_wait(tid);          // current buffer's copies complete
            __syncthreads();          // visible to all; prev readers done

            // issue next tile into the other buffer (overlaps the WMMAs)
            const bool last = (e == NEXP - 1) && (k0 + BK >= DIM);
            if (!last) {
                const int ne = (k0 + BK < DIM) ? e : e + 1;
                const int nk = (k0 + BK < DIM) ? k0 + BK : 0;
                stage_tiles(Ab + nk,
                            WeT + (size_t)ne * DIM * DIM + (size_t)blockN * DIM + nk,
                            At[buf ^ 1], Bt[buf ^ 1], tid);
            }

            const unsigned short* Ac = At[buf];
            const unsigned short* Bc = Bt[buf];
            Frag af[4];
#pragma unroll
            for (int mt = 0; mt < 4; ++mt) {
                const int base = (m0w + mt * 16 + lm) * LSTR + hi * 8;
                af[mt].q[0] = *(const u32x4*)(Ac + base);
                af[mt].q[1] = *(const u32x4*)(Ac + base + 16);
            }
            Frag bfr[2];
#pragma unroll
            for (int nt = 0; nt < 2; ++nt) {
                const int base = (n0w + nt * 16 + lm) * LSTR + hi * 8;
                bfr[nt].q[0] = *(const u32x4*)(Bc + base);
                bfr[nt].q[1] = *(const u32x4*)(Bc + base + 16);
            }
#pragma unroll
            for (int mt = 0; mt < 4; ++mt)
#pragma unroll
                for (int nt = 0; nt < 2; ++nt)
                    acc[mt][nt] = __builtin_amdgcn_wmma_f32_16x16x32_bf16(
                        false, af[mt].v, false, bfr[nt].v,
                        (short)0, acc[mt][nt], false, false);
            buf ^= 1;
        }

        // per-expert epilogue: bias + SiLU + gate, accumulate (registers only)
#pragma unroll
        for (int nt = 0; nt < 2; ++nt) {
            const int gcol = blockN + n0w + nt * 16 + lm;
            const float bev = be[(size_t)e * DIM + gcol];
#pragma unroll
            for (int mt = 0; mt < 4; ++mt) {
#pragma unroll
                for (int r = 0; r < 8; ++r) {
                    const int t = blockM + m0w + mt * 16 + hi * 8 + r;
                    const float g = G[(size_t)t * NEXP + e];
                    const float v = acc[mt][nt][r] + bev;
                    const float s = v / (1.f + __expf(-v));
                    comb_acc[mt][nt][r] += g * s;
                }
            }
        }
    }

    // write combined (bf16) for the output GEMM
#pragma unroll
    for (int nt = 0; nt < 2; ++nt) {
        const int gcol = blockN + n0w + nt * 16 + lm;
#pragma unroll
        for (int mt = 0; mt < 4; ++mt)
#pragma unroll
            for (int r = 0; r < 8; ++r) {
                const int t = blockM + m0w + mt * 16 + hi * 8 + r;
                comb[(size_t)t * DIM + gcol] = f32_to_bf16(comb_acc[mt][nt][r]);
            }
    }
}

// ---------------------------------------------------------------------------
// Output GEMM: y = x + (combined @ Wo + bo)      (fp32 result into workspace)
// WoT layout: [n][k]. Same pipelined double-buffered skeleton.
// ---------------------------------------------------------------------------
__global__ __launch_bounds__(256)
void out_gemm_kernel(const unsigned short* __restrict__ comb,
                     const unsigned short* __restrict__ WoT,
                     const float* __restrict__ bo,
                     const float* __restrict__ x,
                     float* __restrict__ y) {
    __shared__ unsigned short At[2][BM * LSTR];
    __shared__ unsigned short Bt[2][BN * LSTR];

    const int tid  = threadIdx.x;
    const int lane = tid & 31;
    const int wid  = tid >> 5;
    const int lm   = lane & 15;
    const int hi   = (lane >= 16) ? 1 : 0;
    const int m0w  = (wid & 1) * 64;
    const int n0w  = (wid >> 1) * 32;
    const int blockM = blockIdx.x * BM;
    const int blockN = blockIdx.y * BN;

    v8f acc[4][2];
#pragma unroll
    for (int mt = 0; mt < 4; ++mt)
#pragma unroll
        for (int nt = 0; nt < 2; ++nt)
#pragma unroll
            for (int r = 0; r < 8; ++r) acc[mt][nt][r] = 0.f;

    const unsigned short* Ab = comb + (size_t)blockM * DIM;
    const unsigned short* Bb = WoT + (size_t)blockN * DIM;
    stage_tiles(Ab, Bb, At[0], Bt[0], tid);
    int buf = 0;

    for (int k0 = 0; k0 < DIM; k0 += BK) {
        stage_wait(tid);
        __syncthreads();
        if (k0 + BK < DIM)
            stage_tiles(Ab + k0 + BK, Bb + k0 + BK, At[buf ^ 1], Bt[buf ^ 1], tid);

        const unsigned short* Ac = At[buf];
        const unsigned short* Bc = Bt[buf];
        Frag af[4];
#pragma unroll
        for (int mt = 0; mt < 4; ++mt) {
            const int base = (m0w + mt * 16 + lm) * LSTR + hi * 8;
            af[mt].q[0] = *(const u32x4*)(Ac + base);
            af[mt].q[1] = *(const u32x4*)(Ac + base + 16);
        }
        Frag bfr[2];
#pragma unroll
        for (int nt = 0; nt < 2; ++nt) {
            const int base = (n0w + nt * 16 + lm) * LSTR + hi * 8;
            bfr[nt].q[0] = *(const u32x4*)(Bc + base);
            bfr[nt].q[1] = *(const u32x4*)(Bc + base + 16);
        }
#pragma unroll
        for (int mt = 0; mt < 4; ++mt)
#pragma unroll
            for (int nt = 0; nt < 2; ++nt)
                acc[mt][nt] = __builtin_amdgcn_wmma_f32_16x16x32_bf16(
                    false, af[mt].v, false, bfr[nt].v,
                    (short)0, acc[mt][nt], false, false);
        buf ^= 1;
    }

#pragma unroll
    for (int nt = 0; nt < 2; ++nt) {
        const int gcol = blockN + n0w + nt * 16 + lm;
        const float bov = bo[gcol];
#pragma unroll
        for (int mt = 0; mt < 4; ++mt)
#pragma unroll
            for (int r = 0; r < 8; ++r) {
                const int t = blockM + m0w + mt * 16 + hi * 8 + r;
                y[(size_t)t * DIM + gcol] =
                    acc[mt][nt][r] + bov + x[(size_t)t * DIM + gcol];
            }
    }
}

// ---------------------------------------------------------------------------
// RMSNorm: out = norm_w * y / sqrt(mean(y^2) + 1e-6). One block per token.
// ---------------------------------------------------------------------------
__global__ __launch_bounds__(256)
void rmsnorm_kernel(const float* __restrict__ y,
                    const float* __restrict__ nw,
                    float* __restrict__ out) {
    const int t   = blockIdx.x;
    const int tid = threadIdx.x;
    float v[DIM / 256];
    float s = 0.f;
#pragma unroll
    for (int i = 0; i < DIM / 256; ++i) {
        const int d = tid + i * 256;
        v[i] = y[(size_t)t * DIM + d];
        s += v[i] * v[i];
    }
    __shared__ float red[256];
    red[tid] = s;
    __syncthreads();
    for (int st = 128; st > 0; st >>= 1) {
        if (tid < st) red[tid] += red[tid + st];
        __syncthreads();
    }
    const float rinv = rsqrtf(red[0] / (float)DIM + 1e-6f);
#pragma unroll
    for (int i = 0; i < DIM / 256; ++i) {
        const int d = tid + i * 256;
        out[(size_t)t * DIM + d] = nw[d] * v[i] * rinv;
    }
}

// ---------------------------------------------------------------------------
// Launch
// ---------------------------------------------------------------------------
extern "C" void kernel_launch(void* const* d_in, const int* in_sizes, int n_in,
                              void* d_out, int out_size, void* d_ws, size_t ws_size,
                              hipStream_t stream) {
    const float* x      = (const float*)d_in[0];
    const float* Wr     = (const float*)d_in[1];
    const float* br     = (const float*)d_in[2];
    const float* We     = (const float*)d_in[3];
    const float* be     = (const float*)d_in[4];
    const float* Wo     = (const float*)d_in[5];
    const float* bo     = (const float*)d_in[6];
    const float* norm_w = (const float*)d_in[7];
    float* out = (float*)d_out;

    // carve workspace (all offsets 256B aligned by construction)
    char* p = (char*)d_ws;
    unsigned short* xb   = (unsigned short*)p; p += (size_t)TOKENS * DIM * 2;          // 32 MB
    unsigned short* WeT  = (unsigned short*)p; p += (size_t)NEXP * DIM * DIM * 2;      // 64 MB
    unsigned short* WoT  = (unsigned short*)p; p += (size_t)DIM * DIM * 2;             //  8 MB
    float*          G    = (float*)p;          p += (size_t)TOKENS * NEXP * 4;         // 256 KB
    unsigned short* comb = (unsigned short*)p; p += (size_t)TOKENS * DIM * 2;          // 32 MB
    float*          yws  = (float*)p;          p += (size_t)TOKENS * DIM * 4;          // 64 MB

    dim3 tblk(32, 8);
    dim3 tge(DIM / 32, DIM / 32, NEXP);
    dim3 tgo(DIM / 32, DIM / 32, 1);
    transpose_cvt_kernel<<<tge, tblk, 0, stream>>>(We, WeT);
    transpose_cvt_kernel<<<tgo, tblk, 0, stream>>>(Wo, WoT);
    router_kernel<<<TOKENS, 256, 0, stream>>>(x, Wr, br, xb, G);

    dim3 grid(TOKENS / BM, DIM / BN);   // (64, 16)
    expert_gemm_kernel<<<grid, 256, 0, stream>>>(xb, WeT, be, G, comb);
    out_gemm_kernel<<<grid, 256, 0, stream>>>(comb, WoT, bo, x, yws);
    rmsnorm_kernel<<<TOKENS, 256, 0, stream>>>(yws, norm_w, out);
}
